// CriticNetwork_41102837022787
// MI455X (gfx1250) — 16-bit and 8-bit matrix operand order repaired and checked by reading, compiled, not run on hardware
//
#include <hip/hip_runtime.h>
#include <cstdint>
#include <cstddef>

constexpr int Bn  = 1024;
constexpr int Tn  = 256;
constexpr int DP  = 64;
constexpr int An  = 8;
constexpr int Hn  = 256;
constexpr int HID = 256;
constexpr int Cn  = 2;
constexpr int Fn  = DP + 1;
constexpr int KX8 = 128;
constexpr int NG  = 3 * Hn;
constexpr int DIN = Hn + An + 1;
constexpr int KD  = 288;
constexpr int BM  = 16;

constexpr int LDS_XQ  = 0;
constexpr int LDS_HQ  = 2048;
constexpr int LDS_WI  = 6144;
constexpr int LDS_SZ  = LDS_WI + NG * KX8;

#if __has_builtin(__builtin_amdgcn_global_load_async_to_lds_b128) && \
    __has_builtin(__builtin_amdgcn_s_wait_asynccnt)
#define ASYNC_X 1
#else
#define ASYNC_X 0
#endif

typedef __attribute__((ext_vector_type(16))) __bf16 v16bf;
typedef __attribute__((ext_vector_type(16))) int    v16i;
typedef __attribute__((ext_vector_type(8)))  float  v8f;
typedef __attribute__((__vector_size__(16))) int    v4i_;

union FragBF { v16bf v; uint4 q[2]; };
union FragQ  { v16i  v; uint2 d[8]; uint4 q[4]; };

__device__ __forceinline__ unsigned short f2bf(float f) {
  unsigned u = __float_as_uint(f);
  unsigned rnd = ((u >> 16) & 1u) + 0x7FFFu;
  return (unsigned short)((u + rnd) >> 16);
}

__device__ __forceinline__ unsigned char f2e4m3(float f) {
  unsigned su = (__float_as_uint(f) >> 31) << 7;
  float a = fabsf(f);
  if (a >= 448.0f) return (unsigned char)(su | 0x7E);
  if (a < 0.015625f) {
    int m = (int)(a * 512.0f + 0.5f);
    return (unsigned char)(su | (unsigned)m);
  }
  unsigned u = __float_as_uint(a);
  u += ((u >> 20) & 1u) + 0x7FFFFu;
  int e = (int)(u >> 23) - 127;
  int code = ((e + 7) << 3) | (int)((u >> 20) & 7u);
  if (code > 0x7E) code = 0x7E;
  return (unsigned char)(su | (unsigned)code);
}

__device__ __forceinline__ void store_fp8_pair(unsigned char* p0, unsigned char* p1,
                                               float a, float b) {
#if __has_builtin(__builtin_amdgcn_cvt_pk_fp8_f32)
  int p = __builtin_amdgcn_cvt_pk_fp8_f32(a, b, 0, false);
  *p0 = (unsigned char)(p & 0xff);
  *p1 = (unsigned char)((p >> 8) & 0xff);
#else
  *p0 = f2e4m3(a);
  *p1 = f2e4m3(b);
#endif
}

__device__ __forceinline__ float fast_exp(float x) {
  return __builtin_amdgcn_exp2f(x * 1.442695041f);
}
__device__ __forceinline__ float fast_sigm(float x) {
  return __builtin_amdgcn_rcpf(1.0f + fast_exp(-x));
}
__device__ __forceinline__ float fast_tanh(float x) {
#if __has_builtin(__builtin_amdgcn_tanhf)
  return __builtin_amdgcn_tanhf(x);
#else
  return 1.0f - 2.0f * __builtin_amdgcn_rcpf(1.0f + __builtin_amdgcn_exp2f(x * 2.885390082f));
#endif
}

__device__ __forceinline__ v16bf loadA16(const unsigned short* p) {
  FragBF f;
  f.q[0] = *reinterpret_cast<const uint4*>(p);
  f.q[1] = *reinterpret_cast<const uint4*>(p + 16);
  return f.v;
}
__device__ __forceinline__ v16bf loadB16(const unsigned short* p) {
  FragBF f;
  f.q[0] = *reinterpret_cast<const uint4*>(p);
  f.q[1] = *reinterpret_cast<const uint4*>(p + 16);
  return f.v;
}
__device__ __forceinline__ v8f wmma_bf16(v16bf a, v16bf b, v8f c) {
  return __builtin_amdgcn_wmma_f32_16x16x32_bf16(false, a, false, b, (short)0, c, false, false);
}

__device__ __forceinline__ v16i loadA8(const unsigned char* row, int kb) {
  FragQ f;
#pragma unroll
  for (int c = 0; c < 2; ++c) {
    const unsigned char* p = row + c * 64 + kb;
    f.q[c * 2 + 0] = *reinterpret_cast<const uint4*>(p);
    f.q[c * 2 + 1] = *reinterpret_cast<const uint4*>(p + 32);
  }
  return f.v;
}
__device__ __forceinline__ v16i loadB8(const unsigned char* col, int kb2) {
  FragQ f;
  f.q[0] = *reinterpret_cast<const uint4*>(col + kb2);
  f.q[1] = *reinterpret_cast<const uint4*>(col + kb2 + 32);
  f.q[2] = *reinterpret_cast<const uint4*>(col + kb2 + 64);
  f.q[3] = *reinterpret_cast<const uint4*>(col + kb2 + 96);
  return f.v;
}
__device__ __forceinline__ v8f wmma_fp8(v16i a, v16i b, v8f c) {
  return __builtin_amdgcn_wmma_f32_16x16x128_fp8_fp8(a, b, (short)0, c, false, false);
}

__global__ void prep_x8(const float* __restrict__ particles,
                        const float* __restrict__ pweights,
                        unsigned char* __restrict__ xall) {
  size_t i = (size_t)blockIdx.x * 256 + threadIdx.x;
  if (i >= (size_t)Bn * Tn * KX8) return;
  int k = (int)(i & (size_t)(KX8 - 1));
  size_t bt = i >> 7;
  int t = (int)(bt & (size_t)(Tn - 1));
  int b = (int)(bt >> 8);
  float v;
  if (k < DP)       v = particles[bt * DP + k];
  else if (k == DP) v = pweights[bt];
  else              v = 0.0f;
  int blk = b >> 4, row = b & 15;
  xall[(((size_t)blk * Tn + t) * BM + row) * KX8 + k] = f2e4m3(v);
}
__global__ void prep_wi8(const float* __restrict__ Wi, unsigned char* __restrict__ WiT8) {
  int i = blockIdx.x * 256 + threadIdx.x;
  if (i >= NG * KX8) return;
  int n = i / KX8, k = i % KX8;
  WiT8[i] = (k < Fn) ? f2e4m3(Wi[(size_t)k * NG + n]) : (unsigned char)0;
}
__global__ void prep_wh8(const float* __restrict__ Wh, unsigned char* __restrict__ WhT8) {
  int i = blockIdx.x * 256 + threadIdx.x;
  if (i >= NG * Hn) return;
  int n = i / Hn, k = i % Hn;
  WhT8[i] = f2e4m3(Wh[(size_t)k * NG + n]);
}
__global__ void prep_w1T(const float* __restrict__ W1, unsigned short* __restrict__ W1T) {
  int i = blockIdx.x * 256 + threadIdx.x;
  if (i >= Cn * HID * KD) return;
  int c = i / (HID * KD), r = i % (HID * KD);
  int n = r / KD, k = r % KD;
  float v = (k < DIN) ? W1[((size_t)c * DIN + k) * HID + n] : 0.0f;
  W1T[i] = f2bf(v);
}
__global__ void prep_w2T(const float* __restrict__ W2, unsigned short* __restrict__ W2T) {
  int i = blockIdx.x * 256 + threadIdx.x;
  if (i >= Cn * HID * Hn) return;
  int c = i / (HID * Hn), r = i % (HID * Hn);
  int n = r / Hn, k = r % Hn;
  W2T[i] = f2bf(W2[((size_t)c * Hn + k) * HID + n]);
}

__global__ __launch_bounds__(512, 1) void gru_scan(
    const unsigned char* __restrict__ xall,
    const float* __restrict__ bi, const float* __restrict__ bhn,
    const unsigned char* __restrict__ WiT8, const unsigned char* __restrict__ WhT8,
    float* __restrict__ hbuf) {
  extern __shared__ __align__(16) unsigned char smem[];
  unsigned char* xq    = smem + LDS_XQ;
  unsigned char* hq    = smem + LDS_HQ;
  unsigned char* wiLds = smem + LDS_WI;

  const int tid = threadIdx.x;
  const int blk = blockIdx.x;
  const int m0  = blk * BM;
  const unsigned char* xblk = xall + (size_t)blk * Tn * (BM * KX8);

  for (int i = tid; i < BM * Hn; i += 512) hq[i] = (unsigned char)0;
  for (int i = tid * 16; i < NG * KX8; i += 512 * 16)
    *reinterpret_cast<uint4*>(wiLds + i) = *reinterpret_cast<const uint4*>(WiT8 + i);

  auto issueX = [&](int t) {
    if (tid < 128) {
      const unsigned char* g = xblk + (size_t)t * (BM * KX8) + tid * 16;
      unsigned char* l = &xq[tid * 16];
#if ASYNC_X
      __builtin_amdgcn_global_load_async_to_lds_b128(
          (__attribute__((address_space(1))) v4i_*)g,
          (__attribute__((address_space(3))) v4i_*)l, 0, 0);
#else
      *reinterpret_cast<uint4*>(l) = *reinterpret_cast<const uint4*>(g);
#endif
    }
  };
  auto waitX = [&]() {
#if ASYNC_X
    __builtin_amdgcn_s_wait_asynccnt(0);
#endif
  };

  issueX(0);

  const int lane = tid & 31;
  const int wave = tid >> 5;
  const int j0   = wave * 16;
  const int arow = lane & 15;
  const int kbA  = (lane >> 4) * 16;
  const int ncol = lane & 15;
  const int kbB  = (lane >> 4) * 16;
  const int crow = (lane >> 4) * 8;
  const int j    = j0 + ncol;

  const int nr = j;
  const int nz = 256 + j;
  const int nn = 512 + j;
  const v16i wh_r0 = loadB8(WhT8 + (size_t)nr * Hn,       kbB);
  const v16i wh_r1 = loadB8(WhT8 + (size_t)nr * Hn + 128, kbB);
  const v16i wh_z0 = loadB8(WhT8 + (size_t)nz * Hn,       kbB);
  const v16i wh_z1 = loadB8(WhT8 + (size_t)nz * Hn + 128, kbB);
  const v16i wh_n0 = loadB8(WhT8 + (size_t)nn * Hn,       kbB);
  const v16i wh_n1 = loadB8(WhT8 + (size_t)nn * Hn + 128, kbB);

  const float bR  = bi[nr];
  const float bZ  = bi[nz];
  const float bXn = bi[nn];
  const float bHn = bhn[j];

  float hreg[8];
#pragma unroll
  for (int r = 0; r < 8; ++r) hreg[r] = 0.0f;

  const v8f vzero = {0.f,0.f,0.f,0.f,0.f,0.f,0.f,0.f};

  waitX();
  __syncthreads();

  for (int t = 0; t < Tn; ++t) {
    v16i ax  = loadA8(&xq[arow * KX8], kbA);
    v16i ah0 = loadA8(&hq[arow * Hn], kbA);
    v16i ah1 = loadA8(&hq[arow * Hn + 128], kbA);
    __syncthreads();

    if (t + 1 < Tn) issueX(t + 1);

    v8f ar = wmma_fp8(ax, loadB8(wiLds + (size_t)nr * KX8, kbB), vzero);
    ar     = wmma_fp8(ah0, wh_r0, ar);
    ar     = wmma_fp8(ah1, wh_r1, ar);
    v8f az = wmma_fp8(ax, loadB8(wiLds + (size_t)nz * KX8, kbB), vzero);
    az     = wmma_fp8(ah0, wh_z0, az);
    az     = wmma_fp8(ah1, wh_z1, az);
    v8f axn = wmma_fp8(ax, loadB8(wiLds + (size_t)nn * KX8, kbB), vzero);
    v8f ahn = wmma_fp8(ah0, wh_n0, vzero);
    ahn     = wmma_fp8(ah1, wh_n1, ahn);

#pragma unroll
    for (int r = 0; r < 8; ++r) {
      float R  = fast_sigm(ar[r] + bR);
      float Z  = fast_sigm(az[r] + bZ);
      float Nn = fast_tanh(axn[r] + bXn + R * (ahn[r] + bHn));
      hreg[r] = (1.0f - Z) * Nn + Z * hreg[r];
    }
#pragma unroll
    for (int r = 0; r < 8; r += 2) {
      store_fp8_pair(&hq[(crow + r) * Hn + j], &hq[(crow + r + 1) * Hn + j],
                     hreg[r], hreg[r + 1]);
    }
    waitX();
    __syncthreads();
  }

#pragma unroll
  for (int r = 0; r < 8; ++r)
    hbuf[(size_t)(m0 + crow + r) * Hn + j] = hreg[r];
}

__global__ __launch_bounds__(256) void critic_mlp(
    const float* __restrict__ hbuf, const float* __restrict__ action,
    const float* __restrict__ timeidx,
    const unsigned short* __restrict__ W1T, const float* __restrict__ b1,
    const unsigned short* __restrict__ W2T, const float* __restrict__ b2,
    const float* __restrict__ W3, const float* __restrict__ b3,
    float* __restrict__ out) {
  __shared__ __align__(16) unsigned short xb[BM * KD];
  __shared__ __align__(16) unsigned short h1[BM * HID];
  __shared__ float h2[BM * HID];
  __shared__ float b1s[HID], b2s[HID];

  const int tid = threadIdx.x;
  const int c   = blockIdx.y;
  const int m0  = blockIdx.x * BM;

  for (int i = tid; i < HID; i += 256) { b1s[i] = b1[c * HID + i]; b2s[i] = b2[c * HID + i]; }
  for (int i = tid; i < BM * KD; i += 256) {
    int row = i / KD, k = i % KD;
    int b = m0 + row;
    float v;
    if (k < Hn)            v = hbuf[(size_t)b * Hn + k];
    else if (k < Hn + An)  v = action[(size_t)b * An + (k - Hn)];
    else if (k == Hn + An) v = timeidx[b] * (1.0f / 100.0f);
    else                   v = 0.0f;
    xb[i] = f2bf(v);
  }
  __syncthreads();

  const int lane = tid & 31;
  const int wave = tid >> 5;
  const int n0   = wave * 32;
  const int arow = lane & 15;
  const int kbA  = (lane >> 4) * 8;
  const int ncol = lane & 15;
  const int kbB  = (lane >> 4) * 8;
  const int crow = (lane >> 4) * 8;

  v16bf a1[9];
#pragma unroll
  for (int kc = 0; kc < 9; ++kc) a1[kc] = loadA16(&xb[arow * KD + kc * 32 + kbA]);
#pragma unroll
  for (int ti = 0; ti < 2; ++ti) {
    int nt = n0 + ti * 16;
    v8f acc = {0.f,0.f,0.f,0.f,0.f,0.f,0.f,0.f};
#pragma unroll
    for (int kc = 0; kc < 9; ++kc)
      acc = wmma_bf16(a1[kc], loadB16(W1T + ((size_t)c * HID + nt + ncol) * KD + kc * 32 + kbB), acc);
#pragma unroll
    for (int r = 0; r < 8; ++r) {
      int m = crow + r, n = nt + ncol;
      float v = acc[r] + b1s[n];
      h1[m * HID + n] = f2bf(v > 0.0f ? v : 0.0f);
    }
  }
  __syncthreads();

  v16bf a2[8];
#pragma unroll
  for (int kc = 0; kc < 8; ++kc) a2[kc] = loadA16(&h1[arow * HID + kc * 32 + kbA]);
#pragma unroll
  for (int ti = 0; ti < 2; ++ti) {
    int nt = n0 + ti * 16;
    v8f acc = {0.f,0.f,0.f,0.f,0.f,0.f,0.f,0.f};
#pragma unroll
    for (int kc = 0; kc < 8; ++kc)
      acc = wmma_bf16(a2[kc], loadB16(W2T + ((size_t)c * HID + nt + ncol) * Hn + kc * 32 + kbB), acc);
#pragma unroll
    for (int r = 0; r < 8; ++r) {
      int m = crow + r, n = nt + ncol;
      float v = acc[r] + b2s[n];
      h2[m * HID + n] = v > 0.0f ? v : 0.0f;
    }
  }
  __syncthreads();

  if (tid < BM) {
    const float* w3 = W3 + (size_t)c * HID;
    float s = 0.0f;
    for (int k = 0; k < HID; ++k) s += h2[tid * HID + k] * w3[k];
    out[(size_t)(m0 + tid) * Cn + c] = s + b3[c];
  }
}

extern "C" void kernel_launch(void* const* d_in, const int* in_sizes, int n_in,
                              void* d_out, int out_size, void* d_ws, size_t ws_size,
                              hipStream_t stream) {
  (void)in_sizes; (void)n_in; (void)out_size; (void)ws_size;
  const float* particles = (const float*)d_in[0];
  const float* pweights  = (const float*)d_in[1];
  const float* action    = (const float*)d_in[2];
  const float* time_idx  = (const float*)d_in[3];
  const float* Wi        = (const float*)d_in[4];
  const float* bi        = (const float*)d_in[5];
  const float* Wh        = (const float*)d_in[6];
  const float* bhn       = (const float*)d_in[7];
  const float* W1        = (const float*)d_in[8];
  const float* b1        = (const float*)d_in[9];
  const float* W2        = (const float*)d_in[10];
  const float* b2        = (const float*)d_in[11];
  const float* W3        = (const float*)d_in[12];
  const float* b3        = (const float*)d_in[13];
  float* out = (float*)d_out;

  char* ws = (char*)d_ws;
  float*          hbuf = (float*)(ws + 0);
  unsigned char*  WiT8 = (unsigned char*)(ws + 1048576);
  unsigned char*  WhT8 = (unsigned char*)(ws + 1146880);
  unsigned short* W1T  = (unsigned short*)(ws + 1343488);
  unsigned short* W2T  = (unsigned short*)(ws + 1638400);
  unsigned char*  xall = (unsigned char*)(ws + 1900544);

  prep_x8<<<(int)(((size_t)Bn * Tn * KX8 + 255) / 256), 256, 0, stream>>>(particles, pweights, xall);
  prep_wi8<<<(NG * KX8 + 255) / 256, 256, 0, stream>>>(Wi, WiT8);
  prep_wh8<<<(NG * Hn + 255) / 256, 256, 0, stream>>>(Wh, WhT8);
  prep_w1T<<<(Cn * HID * KD + 255) / 256, 256, 0, stream>>>(W1, W1T);
  prep_w2T<<<(Cn * HID * Hn + 255) / 256, 256, 0, stream>>>(W2, W2T);

  gru_scan<<<Bn / BM, 512, LDS_SZ, stream>>>(xall, bi, bhn, WiT8, WhT8, hbuf);
  critic_mlp<<<dim3(Bn / BM, Cn), 256, 0, stream>>>(hbuf, action, time_idx,
                                                    W1T, b1, W2T, b2, W3, b3, out);
}
